// BiRWKV_41815801594115
// MI455X (gfx1250) — matrix operand order repaired and checked by reading, not compiled, not run on hardware
//
#include <hip/hip_runtime.h>

// ---------------------------------------------------------------------------
// BiRWKV-6 (fwd+bwd) for MI455X / gfx1250, wave32.
// GEMMs: v_wmma_f32_16x16x32_bf16 with TDM (tensor_load_to_lds) tile staging,
// double-buffered LDS, per-wave quarter-tile DMA, s_wait_tensorcnt sync.
// ---------------------------------------------------------------------------

typedef __attribute__((ext_vector_type(16))) __bf16 bf16x16;
typedef __attribute__((ext_vector_type(8)))  __bf16 bf16x8;
typedef __attribute__((ext_vector_type(8)))  float  f32x8;
typedef __attribute__((ext_vector_type(4)))  unsigned int u32x4v;
typedef __attribute__((ext_vector_type(8)))  int i32x8v;
typedef __attribute__((ext_vector_type(4)))  int i32x4v;

#define TM 64
#define TN 64
#define TK 32

// Padded LDS tile strides (elements). TDM pad: A rows 16 DW + 8 DW pad = 96 B,
// B rows 32 DW + 8 DW pad = 160 B. Both 16B-aligned, bank-staggered.
#define AS_STRIDE 48
#define BS_STRIDE 80

// group1 word0: data_size=2B (code 1) | pad_enable | pad_interval | pad_amount(8DW->7)
#define TDM_CFG_A ((1u << 16) | (1u << 20) | (3u << 22) | (7u << 25)) // interval 16 DW
#define TDM_CFG_B ((1u << 16) | (1u << 20) | (4u << 22) | (7u << 25)) // interval 32 DW

// ---------------------------------------------------------------------------
// One 2D TDM tile load: global (tile_y rows x tile_x elems, row stride
// stride_x elems, 2B/elem) -> LDS at lds_addr (with row padding per cfg).
// Descriptor packing per CDNA5 ISA.
// This toolchain exposes the 6-arg builtin: (g0, g1, g2, g3, g4, cpol).
// ---------------------------------------------------------------------------
__device__ __forceinline__ void tdm_load_tile(
    unsigned lds_addr, const void* gptr,
    unsigned tile_x, unsigned tile_y,
    unsigned tensor_x, unsigned tensor_y,
    unsigned stride_x, unsigned cfg_word)
{
    const unsigned long long ga = (unsigned long long)gptr;
    u32x4v g0;
    g0[0] = 1u;                                               // count=1, user desc
    g0[1] = lds_addr;                                         // lds_addr[31:0]
    g0[2] = (unsigned)ga;                                     // global_addr[31:0]
    g0[3] = (unsigned)((ga >> 32) & 0x01FFFFFFull) | (2u << 30); // addr[56:32]|type=2

    i32x8v g1;
    g1[0] = (int)cfg_word;                                    // mask=0,dsize,pad cfg
    g1[1] = (int)((tensor_x & 0xFFFFu) << 16);                // dim0[15:0]
    g1[2] = (int)((tensor_x >> 16) | ((tensor_y & 0xFFFFu) << 16)); // dim0hi|dim1lo
    g1[3] = (int)((tensor_y >> 16) | (tile_x << 16));         // dim1[31:16]|tile0
    g1[4] = (int)(tile_y & 0xFFFFu);                          // tile1 (tile2=0)
    g1[5] = (int)stride_x;                                    // dim0_stride[31:0]
    g1[6] = 0;                                                // stride0hi|stride1lo
    g1[7] = 0;
    i32x4v z4 = {0, 0, 0, 0};
    i32x8v z8 = {0, 0, 0, 0, 0, 0, 0, 0};
    __builtin_amdgcn_tensor_load_to_lds(g0, g1, z4, z4, z8, 0);
}

// ---------------------------------------------------------------------------
// WMMA GEMM: C(f32, MxN) = A(bf16, MxK) @ B(bf16, KxN), row-major.
// 128 threads = 4 waves; block tile 64x64; wave tile 32x32 (2x2 WMMA).
// N is a template constant (immediate-offset C stores). M mult of 64,
// K mult of 32, N in {64, 768}.
// ---------------------------------------------------------------------------
template <int N>
__global__ __launch_bounds__(128) void gemm_bf16_wmma(
    const __bf16* __restrict__ A, const __bf16* __restrict__ B,
    float* __restrict__ C, int M, int K)
{
    __shared__ __align__(16) __bf16 As[2][TM][AS_STRIDE];  // 12 KB
    __shared__ __align__(16) __bf16 Bs[2][TK][BS_STRIDE];  // 10 KB

    const int tid  = threadIdx.x;
    const int lane = tid & 31;
    const int wv   = __builtin_amdgcn_readfirstlane(tid >> 5); // uniform wave id
    const int wm   = (wv >> 1) * 32;
    const int wn   = (wv & 1) * 32;
    const int bM   = blockIdx.y * TM;
    const int bN   = blockIdx.x * TN;

    const unsigned asBase = (unsigned)(unsigned long long)(const void*)&As[0][0][0];
    const unsigned bsBase = (unsigned)(unsigned long long)(const void*)&Bs[0][0][0];

    // Each wave DMAs one quarter tile: waves 0/1 -> A halves, 2/3 -> B halves.
    auto prefetch = [&](int k0, int buf) {
        if (wv < 2) {
            tdm_load_tile(asBase + (unsigned)(buf * TM + wv * 32) * (AS_STRIDE * 2),
                          A + (size_t)(bM + wv * 32) * K + k0,
                          TK, 32, (unsigned)K, (unsigned)M, (unsigned)K, TDM_CFG_A);
        } else {
            tdm_load_tile(bsBase + (unsigned)(buf * TK + (wv - 2) * 16) * (BS_STRIDE * 2),
                          B + (size_t)(k0 + (wv - 2) * 16) * N + bN,
                          TN, 16, (unsigned)N, (unsigned)K, (unsigned)N, TDM_CFG_B);
        }
    };

    f32x8 acc[2][2];
#pragma unroll
    for (int i = 0; i < 2; ++i)
#pragma unroll
        for (int j = 0; j < 2; ++j)
#pragma unroll
            for (int e = 0; e < 8; ++e) acc[i][j][e] = 0.0f;

    prefetch(0, 0);
    int buf = 0;
    for (int k0 = 0; k0 < K; k0 += TK) {
        __builtin_amdgcn_s_wait_tensorcnt((short)0); // my quarter-tile landed
        __syncthreads();                             // everyone's tiles visible
        if (k0 + TK < K) prefetch(k0 + TK, buf ^ 1); // overlap next DMA

        // A fragment (16-bit 16x32 layout): lane<16 rows 0..15 khalf=0,
        // lane>=16 khalf=8; elem e -> k = (e>=8?16:0)+khalf+(e&7): two b128 runs.
        // B fragment: lane&15 = n column, elem e -> k = (e>=8?16:0)+khalf+(e&7).
        const int khalf2 = (lane >> 4) * 8;          // element offset in row
        union Frag { bf16x16 v; bf16x8 h[2]; };
        Frag af[2], bfr[2];
#pragma unroll
        for (int i = 0; i < 2; ++i) {
            const __bf16* ap = &As[buf][wm + i * 16 + (lane & 15)][0];
            af[i].h[0] = *(const bf16x8*)(ap + khalf2);
            af[i].h[1] = *(const bf16x8*)(ap + 16 + khalf2);
            const int bn = wn + i * 16 + (lane & 15);
#pragma unroll
            for (int s = 0; s < 16; ++s)
                bfr[i].v[s] = Bs[buf][(s >= 8 ? 16 : 0) + khalf2 + (s & 7)][bn];
        }
#pragma unroll
        for (int i = 0; i < 2; ++i)
#pragma unroll
            for (int j = 0; j < 2; ++j)
                acc[i][j] = __builtin_amdgcn_wmma_f32_16x16x32_bf16(
                    false, af[i].v, false, bfr[j].v, (short)0, acc[i][j],
                    false, false);
        __syncthreads();                             // done reading buf
        buf ^= 1;
    }

    // C/D layout: VGPR e, lane l -> M = e + 8*(l>>4), N = l&15.
    // Template-constant N => e*N folds into store immediate offsets.
    const int crow = 8 * (lane >> 4);
    const int ccol = lane & 15;
#pragma unroll
    for (int i = 0; i < 2; ++i)
#pragma unroll
        for (int j = 0; j < 2; ++j) {
            float* cp = C + (size_t)(bM + wm + i * 16 + crow) * N
                          + (bN + wn + j * 16 + ccol);
#pragma unroll
            for (int e = 0; e < 8; ++e)
                cp[(size_t)e * N] = acc[i][j][e];
        }
}

// ---------------------------------------------------------------------------
// f32 -> bf16 conversion
// ---------------------------------------------------------------------------
__global__ void cvt_bf16_kernel(const float* __restrict__ src,
                                __bf16* __restrict__ dst, int n)
{
    const int i = blockIdx.x * 256 + threadIdx.x;
    if (i < n) dst[i] = (__bf16)src[i];
}

// ---------------------------------------------------------------------------
// Token shift + 5-way mix, both directions, writes bf16 GEMM operands.
// xmix layout: [dir][i][bt*C + c], i in {r,k,v,w,g}.
// ---------------------------------------------------------------------------
__global__ void mix_kernel(const float* __restrict__ x,
                           const float* __restrict__ mu,
                           __bf16* __restrict__ xmix,
                           int B, int T, int C)
{
    const int idx = blockIdx.x * 256 + threadIdx.x;
    const int total = B * T * C;
    if (idx >= total) return;
    const int c  = idx % C;
    const int bt = idx / C;
    const int t  = bt % T;
    const float xv = x[idx];
    const float xprev = (t > 0)     ? x[idx - C] : 0.0f;
    const float xnext = (t < T - 1) ? x[idx + C] : 0.0f;
#pragma unroll
    for (int dir = 0; dir < 2; ++dir) {
        const float dx = (dir ? xnext : xprev) - xv;
#pragma unroll
        for (int i = 0; i < 5; ++i) {
            const float m = mu[(dir * 5 + i) * C + c];
            xmix[(size_t)(dir * 5 + i) * total + idx] = (__bf16)(xv + dx * m);
        }
    }
}

__global__ void tanh_bf16_kernel(const float* __restrict__ src,
                                 __bf16* __restrict__ dst, int n)
{
    const int i = blockIdx.x * 256 + threadIdx.x;
    if (i < n) dst[i] = (__bf16)tanhf(src[i]);
}

// w = exp(-exp(w_bias + wpre)), in place over (2, BT, C)
__global__ void decay_kernel(float* __restrict__ wbuf,
                             const float* __restrict__ w_bias,
                             int BTC, int C)
{
    const int i = blockIdx.x * 256 + threadIdx.x;
    if (i >= 2 * BTC) return;
    const int dir = i / BTC;
    const int c   = i % C;
    wbuf[i] = expf(-expf(w_bias[dir * C + c] + wbuf[i]));
}

// ---------------------------------------------------------------------------
// Sequential WKV scan. One 64-thread block per (dir, b, h); thread e owns
// state column S[:, e] (64 f32 in VGPRs). r/k/w/u broadcast via LDS.
//   y[e]    = sum_d r[d] * (S[d][e] + u[d]*k[d]*v[e])
//   S[d][e] = w[d]*S[d][e] + k[d]*v[e]
// dir==1 walks t from T-1 down to 0 (== reversed-sequence forward scan).
// ---------------------------------------------------------------------------
__global__ __launch_bounds__(64) void scan_kernel(
    const float* __restrict__ r, const float* __restrict__ k,
    const float* __restrict__ v, const float* __restrict__ w,
    const float* __restrict__ u, float* __restrict__ y,
    int B, int T, int C, int H)
{
    const int e = threadIdx.x;                 // 0..63: state column
    int blk = blockIdx.x;
    const int BH = B * H;
    const int dir = blk / BH; blk %= BH;
    const int b = blk / H, h = blk % H;

    __shared__ float rs[64], ks[64], ws[64], us[64];
    us[e] = u[(dir * H + h) * 64 + e];

    float S[64];
#pragma unroll
    for (int d = 0; d < 64; ++d) S[d] = 0.0f;

    const size_t dirOff = (size_t)dir * B * T * C;
    for (int t = 0; t < T; ++t) {
        const int tt = dir ? (T - 1 - t) : t;
        const size_t base = dirOff + (size_t)(b * T + tt) * C + h * 64;
        rs[e] = r[base + e];
        ks[e] = k[base + e];
        ws[e] = w[base + e];
        const float ve = v[base + e];
        __syncthreads();
        float acc = 0.0f;
#pragma unroll
        for (int d = 0; d < 64; ++d) {
            const float kv = ks[d] * ve;
            acc += rs[d] * fmaf(us[d], kv, S[d]);
            S[d] = fmaf(ws[d], S[d], kv);
        }
        y[base + e] = acc;
        __syncthreads();
    }
}

// ---------------------------------------------------------------------------
// Per-head GroupNorm over D=64, affine, then * silu(g). Emits bf16 operand
// for the Wo GEMM. One 64-thread block per (dir, bt, h).
// ---------------------------------------------------------------------------
__global__ __launch_bounds__(64) void normgate_kernel(
    const float* __restrict__ y, const float* __restrict__ glin,
    const float* __restrict__ ln_w, const float* __restrict__ ln_b,
    __bf16* __restrict__ outb, int BT, int C, int H)
{
    const int d = threadIdx.x;                  // 0..63
    int blk = blockIdx.x;                       // (dir*BT + bt)*H + h
    const int h = blk % H; blk /= H;
    const int bt = blk % BT;
    const int dir = blk / BT;
    const size_t idx = ((size_t)dir * BT + bt) * (size_t)C + h * 64 + d;

    __shared__ float buf[64];
    const float yv = y[idx];
    buf[d] = yv;
    __syncthreads();
    float s1 = 0.0f, s2 = 0.0f;
#pragma unroll
    for (int i = 0; i < 64; ++i) { const float t = buf[i]; s1 += t; s2 += t * t; }
    const float mean = s1 * (1.0f / 64.0f);
    const float var  = s2 * (1.0f / 64.0f) - mean * mean;
    const float rstd = rsqrtf(var + 1e-5f);
    const int c = h * 64 + d;
    const float gl = glin[idx];
    const float g  = gl / (1.0f + expf(-gl));           // silu
    const float yn = (yv - mean) * rstd * ln_w[dir * C + c] + ln_b[dir * C + c];
    outb[idx] = (__bf16)(yn * g);
}

// out = (o_fwd + o_bwd) * pad_mask
__global__ void final_kernel(const float* __restrict__ o,
                             const unsigned char* __restrict__ mask,
                             float* __restrict__ out, int BT, int C)
{
    const int i = blockIdx.x * 256 + threadIdx.x;
    const int total = BT * C;
    if (i >= total) return;
    const int bt = i / C;
    const float mv = mask[bt] ? 1.0f : 0.0f;
    out[i] = (o[i] + o[(size_t)total + i]) * mv;
}

// ---------------------------------------------------------------------------
// Host side
// ---------------------------------------------------------------------------
extern "C" void kernel_launch(void* const* d_in, const int* in_sizes, int n_in,
                              void* d_out, int out_size, void* d_ws, size_t ws_size,
                              hipStream_t stream)
{
    (void)in_sizes; (void)n_in; (void)out_size; (void)ws_size;
    const int B = 4, T = 1024, C = 768, H = 12, LORA = 64;
    const int BT  = B * T;            // 4096
    const int BTC = BT * C;           // 3,145,728
    const int CC  = C * C;

    const float* x      = (const float*)d_in[0];
    const unsigned char* mask = (const unsigned char*)d_in[1];
    const float* mu     = (const float*)d_in[2];
    const float* w_bias = (const float*)d_in[3];
    const float* dw1    = (const float*)d_in[4];
    const float* dw2    = (const float*)d_in[5];
    const float* u      = (const float*)d_in[6];
    const float* W[5]   = { (const float*)d_in[7],  (const float*)d_in[8],
                            (const float*)d_in[9],  (const float*)d_in[10],
                            (const float*)d_in[11] };   // Wr,Wk,Wv,Wg,Wo
    const float* ln_w   = (const float*)d_in[12];
    const float* ln_b   = (const float*)d_in[13];
    float* out = (float*)d_out;

    // Workspace carve-out (256B aligned slices)
    char* p = (char*)d_ws;
    auto alloc = [&](size_t bytes) -> void* {
        void* q = (void*)p;
        p += (bytes + 255) & ~(size_t)255;
        return q;
    };
    __bf16* xmix = (__bf16*)alloc((size_t)2 * 5 * BTC * 2);
    __bf16* Wbf[5];
    for (int i = 0; i < 5; ++i) Wbf[i] = (__bf16*)alloc((size_t)2 * CC * 2);
    __bf16* dw1b = (__bf16*)alloc((size_t)2 * C * LORA * 2);
    __bf16* dw2b = (__bf16*)alloc((size_t)2 * LORA * C * 2);
    float* rbuf  = (float*)alloc((size_t)2 * BTC * 4);
    float* kbuf  = (float*)alloc((size_t)2 * BTC * 4);
    float* vbuf  = (float*)alloc((size_t)2 * BTC * 4);
    float* gbuf  = (float*)alloc((size_t)2 * BTC * 4);
    float* wbuf  = (float*)alloc((size_t)2 * BTC * 4);   // wpre -> decay
    float* ybuf  = (float*)alloc((size_t)2 * BTC * 4);
    float* obuf  = (float*)alloc((size_t)2 * BTC * 4);
    float* l1    = (float*)alloc((size_t)2 * BT * LORA * 4);
    __bf16* l1b  = (__bf16*)alloc((size_t)2 * BT * LORA * 2);
    __bf16* ynb  = (__bf16*)alloc((size_t)2 * BTC * 2);

    auto cvt = [&](const float* s, __bf16* d, int n) {
        cvt_bf16_kernel<<<(n + 255) / 256, 256, 0, stream>>>(s, d, n);
    };
    auto gemmC = [&](const __bf16* A, const __bf16* Bm, float* Cm, int K) {
        dim3 grid(C / TN, BT / TM);
        gemm_bf16_wmma<768><<<grid, 128, 0, stream>>>(A, Bm, Cm, BT, K);
    };
    auto gemmL = [&](const __bf16* A, const __bf16* Bm, float* Cm) {
        dim3 grid(1, BT / TM);
        gemm_bf16_wmma<64><<<grid, 128, 0, stream>>>(A, Bm, Cm, BT, C);
    };

    // 1) weights -> bf16
    for (int i = 0; i < 5; ++i) cvt(W[i], Wbf[i], 2 * CC);
    cvt(dw1, dw1b, 2 * C * LORA);
    cvt(dw2, dw2b, 2 * LORA * C);

    // 2) token shift + mix (both directions)
    mix_kernel<<<(BTC + 255) / 256, 256, 0, stream>>>(x, mu, xmix, B, T, C);

    // 3) projection + LoRA-1 GEMMs
    for (int dir = 0; dir < 2; ++dir) {
        const size_t plane = (size_t)BTC;
        const __bf16* xr = xmix + (size_t)(dir * 5 + 0) * plane;
        const __bf16* xk = xmix + (size_t)(dir * 5 + 1) * plane;
        const __bf16* xv = xmix + (size_t)(dir * 5 + 2) * plane;
        const __bf16* xw = xmix + (size_t)(dir * 5 + 3) * plane;
        const __bf16* xg = xmix + (size_t)(dir * 5 + 4) * plane;
        gemmC(xr, Wbf[0] + (size_t)dir * CC, rbuf + (size_t)dir * BTC, C);
        gemmC(xk, Wbf[1] + (size_t)dir * CC, kbuf + (size_t)dir * BTC, C);
        gemmC(xv, Wbf[2] + (size_t)dir * CC, vbuf + (size_t)dir * BTC, C);
        gemmC(xg, Wbf[3] + (size_t)dir * CC, gbuf + (size_t)dir * BTC, C);
        gemmL(xw, dw1b + (size_t)dir * C * LORA, l1 + (size_t)dir * BT * LORA);
    }

    // 4) tanh on LoRA hidden, 5) LoRA-2 GEMM, 6) decay transform
    tanh_bf16_kernel<<<(2 * BT * LORA + 255) / 256, 256, 0, stream>>>(
        l1, l1b, 2 * BT * LORA);
    for (int dir = 0; dir < 2; ++dir)
        gemmC(l1b + (size_t)dir * BT * LORA, dw2b + (size_t)dir * LORA * C,
              wbuf + (size_t)dir * BTC, LORA);
    decay_kernel<<<(2 * BTC + 255) / 256, 256, 0, stream>>>(wbuf, w_bias, BTC, C);

    // 7) sequential WKV scan (96 independent (dir,b,h) chains)
    scan_kernel<<<2 * B * H, 64, 0, stream>>>(rbuf, kbuf, vbuf, wbuf, u, ybuf,
                                              B, T, C, H);

    // 8) GroupNorm + silu gate -> bf16
    normgate_kernel<<<2 * BT * H, 64, 0, stream>>>(ybuf, gbuf, ln_w, ln_b, ynb,
                                                   BT, C, H);

    // 9) output projection
    for (int dir = 0; dir < 2; ++dir)
        gemmC(ynb + (size_t)dir * BTC, Wbf[4] + (size_t)dir * CC,
              obuf + (size_t)dir * BTC, C);

    // 10) fwd + bwd, pad mask
    final_kernel<<<(BTC + 255) / 256, 256, 0, stream>>>(obuf, mask, out, BT, C);
}
